// CNNBiLSTMCRF_36283883717415
// MI455X (gfx1250) — matrix operand order repaired and checked by reading, not compiled, not run on hardware
//
#include <hip/hip_runtime.h>
#include <hip/hip_bf16.h>

#define B_  32
#define S_  512
#define V_  30000
#define E_  300
#define EP_ 320        // E padded to multiple of 32 for K of bf16 WMMA
#define H_  256
#define T_  20
#define HH_ 128
#define G_  512        // 4*HH
#define SP_ 514        // S + 2 (zero pad rows for conv)
#define BS_ (B_*S_)

typedef __attribute__((ext_vector_type(16))) __bf16 v16bf;
typedef __attribute__((ext_vector_type(8)))  __bf16 v8bf;
typedef __attribute__((ext_vector_type(8)))  float  v8f;

__device__ __forceinline__ __bf16 f2bf(float f){
  union { float f; unsigned u; } a; a.f = f;
  unsigned u = a.u;
  u += 0x7fffu + ((u >> 16) & 1u);          // round-to-nearest-even
  unsigned short hs = (unsigned short)(u >> 16);
  union { unsigned short s; __bf16 h; } b; b.s = hs;
  return b.h;
}

// Build a 16-element bf16 fragment from two contiguous 8-element (16B) chunks.
__device__ __forceinline__ v16bf ld16(const __bf16* p0, const __bf16* p1){
  v8bf lo = *(const v8bf*)p0;
  v8bf hi = *(const v8bf*)p1;
  return __builtin_shufflevector(lo, hi, 0,1,2,3,4,5,6,7,8,9,10,11,12,13,14,15);
}

// ---------------- weight repack (f32 -> bf16, N-major B operands) -----------
__global__ void prep_kernel(const float* conv_w, const float* w_ih_f, const float* w_ih_b,
                            const float* w_hh_f, const float* w_hh_b,
                            __bf16* wc, __bf16* wih, __bf16* whh){
  int idx = blockIdx.x*blockDim.x + threadIdx.x;
  const int NWC  = H_*3*EP_;    // wc[h*(3*EP) + k*EP + e]  (N-major conv weights, e zero-padded)
  const int NWIH = 2*G_*H_;     // wih[dir][g*H + h]  (same layout as source)
  const int NWHH = 2*G_*HH_;    // whh[dir][g*HH + u]
  if (idx < NWC){
    int h = idx / (3*EP_);
    int r = idx % (3*EP_);
    int k = r / EP_, e = r % EP_;
    float v = (e < E_) ? conv_w[(h*3 + k)*E_ + e] : 0.f;
    wc[idx] = f2bf(v);
  } else if (idx < NWC + NWIH){
    int j = idx - NWC;
    int dir = j / (G_*H_);
    int t = j % (G_*H_);
    const float* w = dir ? w_ih_b : w_ih_f;
    wih[j] = f2bf(w[t]);
  } else if (idx < NWC + NWIH + NWHH){
    int j = idx - NWC - NWIH;
    int dir = j / (G_*HH_);
    int t = j % (G_*HH_);
    const float* w = dir ? w_hh_b : w_hh_f;
    whh[j] = f2bf(w[t]);
  }
}

// ---------------- embedding gather + conv zero padding ----------------------
// xp[b][sp][e], sp in [0,513]; sp==0 / sp==513 are zero rows; e in [300,320) zero.
__global__ void embed_kernel(const int* ids, const float* emb, __bf16* xp){
  int idx = blockIdx.x*blockDim.x + threadIdx.x;   // B*SP*EP threads exactly
  int e  = idx % EP_;
  int sp = (idx / EP_) % SP_;
  int b  = idx / (EP_*SP_);
  float v = 0.f;
  if (sp > 0 && sp < SP_-1 && e < E_){
    int tok = ids[b*S_ + sp - 1];
    v = emb[(size_t)tok*E_ + e];
  }
  xp[idx] = f2bf(v);
}

// ---------------- conv as WMMA GEMM: feats[m][h] = sum_{k,e} xp * wc + bias --
__global__ void conv_gemm_kernel(const __bf16* xp, const __bf16* wc,
                                 const float* conv_b, __bf16* feats){
  int lane = threadIdx.x & 31;
  int wave = threadIdx.x >> 5;
  int tile = blockIdx.x*8 + wave;        // 16384 tiles = 1024(mt) x 16(nt)
  int mt = tile >> 4, nt = tile & 15;
  int m0 = mt*16, n0 = nt*16;
  int bb = m0 >> 9;                      // batch (S_ = 512)
  int s0 = m0 & 511;
  int halfk = lane >> 4;
  int r = lane & 15;
  int n = n0 + (lane & 15);
  float bias = conv_b[n];
  v8f acc;
  #pragma unroll
  for (int i = 0; i < 8; ++i) acc[i] = bias;
  for (int kc = 0; kc < 30; ++kc){       // K = 3 taps * 320, chunks of 32
    int tap = kc / 10;
    int e0  = (kc % 10) * 32;
    const __bf16* arow = xp + (size_t)(bb*SP_ + s0 + r + tap)*EP_ + e0 + halfk*8;
    v16bf a = ld16(arow, arow + 16);
    const __bf16* brow = wc + (size_t)n*(3*EP_) + tap*EP_ + e0 + halfk*8;
    v16bf bf = ld16(brow, brow + 16);
    acc = __builtin_amdgcn_wmma_f32_16x16x32_bf16(false, a, false, bf,
                                                  (short)0, acc, false, false);
  }
  int mbase = m0 + halfk*8;
  #pragma unroll
  for (int i = 0; i < 8; ++i)
    feats[(size_t)(mbase+i)*H_ + n] = f2bf(acc[i]);
}

// ---------------- x-projection GEMMs for both LSTM directions ---------------
__global__ void xproj_gemm_kernel(const __bf16* feats, const __bf16* wih,
                                  const float* bi_f, const float* bh_f,
                                  const float* bi_b, const float* bh_b,
                                  float* xproj){
  int lane = threadIdx.x & 31;
  int wave = threadIdx.x >> 5;
  int tile = blockIdx.x*8 + wave;        // 65536 tiles = 2(dir) x 1024(mt) x 32(nt)
  int dir = tile >> 15;
  int rem = tile & 32767;
  int mt = rem >> 5, nt = rem & 31;
  int m0 = mt*16, n0 = nt*16;
  int halfk = lane >> 4;
  int r = lane & 15;
  int g = n0 + (lane & 15);
  float bias = dir ? (bi_b[g] + bh_b[g]) : (bi_f[g] + bh_f[g]);
  v8f acc;
  #pragma unroll
  for (int i = 0; i < 8; ++i) acc[i] = bias;
  const __bf16* wd = wih + (size_t)dir*G_*H_;
  #pragma unroll
  for (int kc = 0; kc < 8; ++kc){        // K = 256
    const __bf16* arow = feats + (size_t)(m0+r)*H_ + kc*32 + halfk*8;
    v16bf a = ld16(arow, arow + 16);
    const __bf16* brow = wd + (size_t)g*H_ + kc*32 + halfk*8;
    v16bf bf = ld16(brow, brow + 16);
    acc = __builtin_amdgcn_wmma_f32_16x16x32_bf16(false, a, false, bf,
                                                  (short)0, acc, false, false);
  }
  float* xo = xproj + (size_t)dir*BS_*G_;
  int mbase = m0 + halfk*8;
  #pragma unroll
  for (int i = 0; i < 8; ++i)
    xo[(size_t)(mbase+i)*G_ + g] = acc[i];
}

// ---------------- persistent LSTM recurrence (w_hh resident in LDS) ---------
// grid = 4: blockIdx = dir*2 + batch_half. 512 threads = 16 waves.
// LDS: whh bf16 [512][128] (128KB) | h bf16 [16][128] | c f32 [16][128] | gates f32 [16][512]
__global__ void __launch_bounds__(512) lstm_kernel(const float* xproj,
                                                   const __bf16* whh_g,
                                                   float* hout){
  extern __shared__ char smem[];
  __bf16* sh_whh = (__bf16*)smem;                          // 131072 B
  __bf16* sh_h   = (__bf16*)(smem + 131072);               //   4096 B
  float*  sh_c   = (float*) (smem + 131072 + 4096);        //   8192 B
  float*  sh_g   = (float*) (smem + 131072 + 4096 + 8192); //  32768 B
  int dir  = blockIdx.x >> 1;
  int half = blockIdx.x & 1;
  int b0 = half * 16;
  int tid = threadIdx.x;

  { // stage w_hh (N-major [g][u]) into LDS with 16B copies
    const uint4* src = (const uint4*)(whh_g + (size_t)dir*G_*HH_);
    uint4* dst = (uint4*)sh_whh;
    for (int i = tid; i < (G_*HH_/8); i += 512) dst[i] = src[i];
  }
  for (int i = tid; i < 16*HH_; i += 512){ sh_c[i] = 0.f; sh_h[i] = f2bf(0.f); }
  __syncthreads();

  const float* xp = xproj + (size_t)dir*BS_*G_;
  float* ho = hout + (size_t)dir*BS_*HH_;
  int lane = tid & 31, wave = tid >> 5;
  int halfk = lane >> 4;
  int r = lane & 15;

  for (int it = 0; it < S_; ++it){
    int s = dir ? (S_-1-it) : it;
    #pragma unroll
    for (int t2 = 0; t2 < 2; ++t2){              // each wave: 2 of 32 n-tiles
      int n0 = (wave + t2*16) * 16;
      int g = n0 + (lane & 15);
      v8f acc;
      #pragma unroll
      for (int i = 0; i < 8; ++i) acc[i] = 0.f;
      #pragma unroll
      for (int kc = 0; kc < 4; ++kc){            // K = HH = 128
        const __bf16* arow = sh_h + r*HH_ + kc*32 + halfk*8;
        v16bf a = ld16(arow, arow + 16);
        const __bf16* brow = sh_whh + (size_t)g*HH_ + kc*32 + halfk*8;
        v16bf bf = ld16(brow, brow + 16);
        acc = __builtin_amdgcn_wmma_f32_16x16x32_bf16(false, a, false, bf,
                                                      (short)0, acc, false, false);
      }
      int gate = g >> 7;                         // 0:i 1:f 2:g 3:o
      #pragma unroll
      for (int i = 0; i < 8; ++i){
        int mrow = halfk*8 + i;
        float v = acc[i] + xp[(size_t)((b0+mrow)*S_ + s)*G_ + g];
        v = (gate == 2) ? tanhf(v) : 1.f/(1.f + expf(-v));
        sh_g[mrow*G_ + g] = v;
      }
    }
    __syncthreads();
    #pragma unroll
    for (int t2 = 0; t2 < 4; ++t2){              // 16*128 state updates
      int idx = t2*512 + tid;
      int bl = idx >> 7, u = idx & 127;
      float iv = sh_g[bl*G_ + u];
      float fv = sh_g[bl*G_ + 128 + u];
      float gv = sh_g[bl*G_ + 256 + u];
      float ov = sh_g[bl*G_ + 384 + u];
      float cc = fv*sh_c[bl*HH_ + u] + iv*gv;
      float hh = ov*tanhf(cc);
      sh_c[bl*HH_ + u] = cc;
      sh_h[bl*HH_ + u] = f2bf(hh);
      ho[(size_t)((b0+bl)*S_ + s)*HH_ + u] = hh;
    }
    __syncthreads();
  }
}

// ---------------- emissions: [B*S][T] = concat(h_f,h_b) @ out_w^T + out_b ---
__global__ void emis_kernel(const float* hbuf, const float* out_w,
                            const float* out_b, float* em){
  __shared__ float row[256];
  int m = blockIdx.x;
  int tid = threadIdx.x;
  if (tid < 128) row[tid] = hbuf[(size_t)m*HH_ + tid];                           // h_f
  else           row[tid] = hbuf[(size_t)BS_*HH_ + (size_t)m*HH_ + (tid - 128)]; // h_b
  __syncthreads();
  if (tid < T_){
    float a = out_b[tid];
    for (int h = 0; h < H_; ++h) a += row[h] * out_w[tid*H_ + h];
    em[(size_t)m*T_ + tid] = a;
  }
}

// ---------------- CRF loss: numerator + forward algorithm + mean ------------
__global__ void __launch_bounds__(1024) crf_kernel(const float* em, const int* labels,
                                                   const int* mask, const float* start_t,
                                                   const float* end_t, const float* trans,
                                                   float* out){
  __shared__ float s_alpha[B_][T_];
  __shared__ float s_trans[T_][T_];
  __shared__ float s_start[T_], s_end[T_];
  __shared__ float s_num[B_], s_denom[B_];
  int tid = threadIdx.x;
  if (tid < T_*T_) s_trans[tid / T_][tid % T_] = trans[tid];
  if (tid < T_){ s_start[tid] = start_t[tid]; s_end[tid] = end_t[tid]; }
  __syncthreads();

  if (tid < B_){
    int b = tid;
    const int* lb = labels + b*S_;
    const int* mk = mask + b*S_;
    const float* eb = em + (size_t)b*S_*T_;
    float num = s_start[lb[0]] + eb[lb[0]];
    int cnt = mk[0];
    for (int s = 1; s < S_; ++s){
      if (mk[s]) num += s_trans[lb[s-1]][lb[s]] + eb[s*T_ + lb[s]];
      cnt += mk[s];
    }
    num += s_end[lb[cnt - 1]];
    s_num[b] = num;
  }

  int b = tid / T_, j = tid % T_;
  bool act = tid < B_*T_;
  if (act) s_alpha[b][j] = s_start[j] + em[(size_t)b*S_*T_ + j];
  __syncthreads();
  for (int s = 1; s < S_; ++s){
    float nxt = 0.f;
    if (act){
      float mx = -1e30f;
      #pragma unroll
      for (int i = 0; i < T_; ++i)
        mx = fmaxf(mx, s_alpha[b][i] + s_trans[i][j]);
      float sum = 0.f;
      #pragma unroll
      for (int i = 0; i < T_; ++i)
        sum += expf(s_alpha[b][i] + s_trans[i][j] - mx);
      nxt = mx + logf(sum) + em[((size_t)b*S_ + s)*T_ + j];
      if (mask[b*S_ + s] == 0) nxt = s_alpha[b][j];
    }
    __syncthreads();
    if (act) s_alpha[b][j] = nxt;
    __syncthreads();
  }
  if (tid < B_){
    float mx = -1e30f;
    for (int j2 = 0; j2 < T_; ++j2) mx = fmaxf(mx, s_alpha[tid][j2] + s_end[j2]);
    float sum = 0.f;
    for (int j2 = 0; j2 < T_; ++j2) sum += expf(s_alpha[tid][j2] + s_end[j2] - mx);
    s_denom[tid] = mx + logf(sum);
  }
  __syncthreads();
  if (tid == 0){
    float acc = 0.f;
    for (int bb = 0; bb < B_; ++bb) acc += s_num[bb] - s_denom[bb];
    out[0] = -acc / (float)B_;
  }
}

extern "C" void kernel_launch(void* const* d_in, const int* in_sizes, int n_in,
                              void* d_out, int out_size, void* d_ws, size_t ws_size,
                              hipStream_t stream){
  const int*   ids     = (const int*)d_in[0];
  const int*   labels  = (const int*)d_in[1];
  const int*   mask    = (const int*)d_in[2];
  const float* emb     = (const float*)d_in[3];
  const float* conv_w  = (const float*)d_in[4];
  const float* conv_b  = (const float*)d_in[5];
  const float* w_ih_f  = (const float*)d_in[6];
  const float* w_hh_f  = (const float*)d_in[7];
  const float* b_ih_f  = (const float*)d_in[8];
  const float* b_hh_f  = (const float*)d_in[9];
  const float* w_ih_b  = (const float*)d_in[10];
  const float* w_hh_b  = (const float*)d_in[11];
  const float* b_ih_b  = (const float*)d_in[12];
  const float* b_hh_b  = (const float*)d_in[13];
  const float* out_w   = (const float*)d_in[14];
  const float* out_b   = (const float*)d_in[15];
  const float* start_t = (const float*)d_in[16];
  const float* end_t   = (const float*)d_in[17];
  const float* trans   = (const float*)d_in[18];

  char* ws = (char*)d_ws;
  __bf16* xp    = (__bf16*)(ws + 0);           // B*SP*EP bf16      (10,526,720 B)
  __bf16* wc    = (__bf16*)(ws + 10526720);    // H*3*EP bf16       (   491,520 B)
  __bf16* wih   = (__bf16*)(ws + 11018240);    // 2*G*H bf16        (   524,288 B)
  __bf16* whh   = (__bf16*)(ws + 11542528);    // 2*G*HH bf16       (   262,144 B)
  __bf16* feats = (__bf16*)(ws + 11804672);    // BS*H bf16         ( 8,388,608 B)
  float*  xproj = (float*) (ws + 20193280);    // 2*BS*G f32        (67,108,864 B)
  float*  hbuf  = (float*) (ws + 87302144);    // 2*BS*HH f32       (16,777,216 B)
  float*  em    = (float*) (ws + 104079360);   // BS*T f32          ( 1,310,720 B)

  prep_kernel<<<2496, 256, 0, stream>>>(conv_w, w_ih_f, w_ih_b, w_hh_f, w_hh_b, wc, wih, whh);
  embed_kernel<<<20560, 256, 0, stream>>>(ids, emb, xp);
  conv_gemm_kernel<<<2048, 256, 0, stream>>>(xp, wc, conv_b, feats);
  xproj_gemm_kernel<<<8192, 256, 0, stream>>>(feats, wih, b_ih_f, b_hh_f, b_ih_b, b_hh_b, xproj);
  lstm_kernel<<<4, 512, 176128, stream>>>(xproj, whh, hbuf);
  emis_kernel<<<16384, 256, 0, stream>>>(hbuf, out_w, out_b, em);
  crf_kernel<<<1, 1024, 0, stream>>>(em, labels, mask, start_t, end_t, trans, (float*)d_out);
}
